// Model_39676907882672
// MI455X (gfx1250) — matrix operand order repaired and checked by reading, not compiled, not run on hardware
//
#include <hip/hip_runtime.h>

typedef __attribute__((ext_vector_type(16))) __bf16 v16bf;
typedef __attribute__((ext_vector_type(8)))  float  v8f;
typedef __attribute__((ext_vector_type(4)))  __bf16 v4bf;

constexpr int Bsz = 32, Hn = 3, Sq = 512, Dd = 512;
constexpr int HD  = Hn * Dd;            // 1536
constexpr int LDSS = 40;                // 32 + 8 bf16 pad -> 80B row stride, conflict-free
constexpr int TILEE = 128 * LDSS;       // bf16 elements per (A or B) tile
constexpr int TILEB = TILEE * 2;        // bytes per tile

// ---------------------------------------------------------------- convert f32 -> bf16
__global__ void cvt_f32_to_bf16(const float* __restrict__ in, __bf16* __restrict__ out, long n4) {
    long i = (long)blockIdx.x * blockDim.x + threadIdx.x;
    long stride = (long)gridDim.x * blockDim.x;
    for (; i < n4; i += stride) {
        float4 f = reinterpret_cast<const float4*>(in)[i];
        v4bf o;
        o[0] = (__bf16)f.x; o[1] = (__bf16)f.y; o[2] = (__bf16)f.z; o[3] = (__bf16)f.w;
        reinterpret_cast<v4bf*>(out)[i] = o;
    }
}

// ---------------------------------------------------------------- column softmax over S axis
// X: [Z, S, D] f32. out = softmax_col(X) * (mul ? mul : 1). In-place safe (thread owns column).
__global__ void col_softmax(const float* __restrict__ X, const float* __restrict__ mul,
                            void* __restrict__ out, int outBf16) {
    int zi = blockIdx.y;
    int d  = blockIdx.x * blockDim.x + threadIdx.x;
    long base = (long)zi * Sq * Dd + d;
    float m = -1e30f, sum = 0.f;
    for (int s = 0; s < Sq; ++s) {
        float x  = X[base + (long)s * Dd];
        float mn = fmaxf(m, x);
        sum = sum * __expf(m - mn) + __expf(x - mn);
        m = mn;
    }
    float inv = 1.f / sum;
    for (int s = 0; s < Sq; ++s) {
        long idx = base + (long)s * Dd;
        float y = __expf(X[idx] - m) * inv;
        if (mul) y *= mul[idx];
        if (outBf16) ((__bf16*)out)[idx] = (__bf16)y;
        else         ((float*)out)[idx]  = y;
    }
}

// ---------------------------------------------------------------- WMMA NT GEMM
// C[M,N] = A[M,K] * B[N,K]^T   (A,B bf16 row-major). MODE: 0=f32 out, 1=bf16 transposed out,
// 2=bf16 out into attention layout [b, s, h*D + e], 3=f32 out + bias.
union FragBF { uint4 u[2]; v16bf v; };

__device__ __forceinline__ void async_b128(unsigned lds_off, unsigned goff, const __bf16* base) {
    // memory -> LDS, no VGPR staging; tracked with ASYNCcnt
    asm volatile("global_load_async_to_lds_b128 %0, %1, %2"
                 :: "v"(lds_off), "v"(goff), "s"(base) : "memory");
}
__device__ __forceinline__ void wait_async0() {
    asm volatile("s_wait_asynccnt 0x0" ::: "memory");
}

template<int MODE>
__launch_bounds__(256)
__global__ void gemm_nt(const __bf16* __restrict__ Ab, const __bf16* __restrict__ Bb,
                        void* __restrict__ Cv, const float* __restrict__ bias,
                        int K, int lda, int ldb, int ldc,
                        long sA, long sB, long sC, int bmod) {
    // double-buffered: [buf0 A | buf0 B | buf1 A | buf1 B]
    __shared__ __align__(16) __bf16 sm[4 * TILEE];

    int zi = blockIdx.z;
    const __bf16* A = Ab + (long)zi * sA;
    const __bf16* B = Bb + (long)(bmod ? (zi % bmod) : zi) * sB;
    long coff;
    if (MODE == 2) { int b = zi / Hn, h = zi % Hn; coff = (long)b * Sq * HD + (long)h * Dd; }
    else           { coff = (long)zi * sC; }

    int t = threadIdx.x;
    int lane = t & 31, wid = t >> 5;
    int waveM = wid & 1, waveN = wid >> 1;          // 2 (M) x 4 (N) waves
    int m0 = blockIdx.y * 128, n0 = blockIdx.x * 128;
    int half = lane >> 4, r16 = lane & 15;

    // per-thread async-copy addressing (two 16B chunks of the 128x32 tile each)
    unsigned smbase = (unsigned)(size_t)(&sm[0]);
    unsigned gA[2], gB[2], lA[2], lB[2];
    #pragma unroll
    for (int j = 0; j < 2; ++j) {
        int i = j * 256 + t;
        int row = i >> 2, c4 = i & 3;
        gA[j] = (unsigned)(((long)(m0 + row) * lda + c4 * 8) * 2);   // byte offset from A
        gB[j] = (unsigned)(((long)(n0 + row) * ldb + c4 * 8) * 2);   // byte offset from B
        lA[j] = smbase + (unsigned)((row * LDSS + c4 * 8) * 2);
        lB[j] = lA[j] + TILEB;
    }

    v8f acc[4][2];
    #pragma unroll
    for (int i = 0; i < 4; ++i)
        #pragma unroll
        for (int j = 0; j < 2; ++j)
            #pragma unroll
            for (int e = 0; e < 8; ++e) acc[i][j][e] = 0.f;

    auto issue = [&](int buf, int k0) {
        unsigned badd = (unsigned)buf * (2 * TILEB);
        unsigned kb = (unsigned)k0 * 2;
        #pragma unroll
        for (int j = 0; j < 2; ++j) {
            async_b128(lA[j] + badd, gA[j] + kb, A);
            async_b128(lB[j] + badd, gB[j] + kb, B);
        }
    };

    int nk = K >> 5;
    issue(0, 0);
    for (int it = 0; it < nk; ++it) {
        wait_async0();
        __syncthreads();                    // all waves' async LDS writes visible
        if (it + 1 < nk) issue((it + 1) & 1, (it + 1) << 5);   // overlap next tile with compute

        const __bf16* As = sm + (size_t)(it & 1) * 2 * TILEE;
        const __bf16* Bs = As + TILEE;

        FragBF fa[4], fb[2];
        #pragma unroll
        for (int mt = 0; mt < 4; ++mt) {
            const __bf16* p = &As[(waveM * 64 + mt * 16 + r16) * LDSS + half * 8];
            fa[mt].u[0] = *reinterpret_cast<const uint4*>(p);       // K chunk [half*8 .. +7]
            fa[mt].u[1] = *reinterpret_cast<const uint4*>(p + 16);  // K chunk [16+half*8 .. +7]
        }
        #pragma unroll
        for (int nt = 0; nt < 2; ++nt) {
            const __bf16* p = &Bs[(waveN * 32 + nt * 16 + r16) * LDSS + half * 8];
            fb[nt].u[0] = *reinterpret_cast<const uint4*>(p);       // K chunk [half*8 .. +7]
            fb[nt].u[1] = *reinterpret_cast<const uint4*>(p + 16);  // K chunk [16+half*8 .. +7]
        }
        #pragma unroll
        for (int mt = 0; mt < 4; ++mt)
            #pragma unroll
            for (int nt = 0; nt < 2; ++nt)
                acc[mt][nt] = __builtin_amdgcn_wmma_f32_16x16x32_bf16(
                    false, fa[mt].v, false, fb[nt].v, (short)0, acc[mt][nt], false, false);
    }

    // epilogue: C fragment -> lane (m = e + 8*half, n = r16)
    #pragma unroll
    for (int mt = 0; mt < 4; ++mt)
        #pragma unroll
        for (int nt = 0; nt < 2; ++nt) {
            int row0 = m0 + waveM * 64 + mt * 16 + half * 8;
            int col  = n0 + waveN * 32 + nt * 16 + r16;
            #pragma unroll
            for (int e = 0; e < 8; ++e) {
                int row = row0 + e;
                float v = acc[mt][nt][e];
                if (MODE == 0)      ((float*)Cv)[coff + (long)row * ldc + col] = v;
                else if (MODE == 1) ((__bf16*)Cv)[coff + (long)col * ldc + row] = (__bf16)v;
                else if (MODE == 2) ((__bf16*)Cv)[coff + (long)row * ldc + col] = (__bf16)v;
                else                ((float*)Cv)[coff + (long)row * ldc + col] = v + bias[col];
            }
        }
}

// ---------------------------------------------------------------- launch
extern "C" void kernel_launch(void* const* d_in, const int* in_sizes, int n_in,
                              void* d_out, int out_size, void* d_ws, size_t ws_size,
                              hipStream_t stream) {
    const float* x1   = (const float*)d_in[0];
    const float* qw   = (const float*)d_in[1];
    const float* kw   = (const float*)d_in[2];
    const float* vw   = (const float*)d_in[3];
    const float* ow   = (const float*)d_in[4];
    const float* bias = (const float*)d_in[5];

    const long nX  = (long)Bsz * Hn * Sq * Dd;   // 25,165,824
    const long nW  = (long)Hn * Dd * Dd;         //    786,432
    const long nOW = (long)HD * HD;              //  2,359,296

    char* ws = (char*)d_ws;
    size_t off = 0;
    auto alloc = [&](size_t bytes) { void* p = ws + off; off = (off + bytes + 255) & ~(size_t)255; return p; };

    __bf16* xb  = (__bf16*)alloc(nX  * 2);
    __bf16* wqb = (__bf16*)alloc(nW  * 2);
    __bf16* wkb = (__bf16*)alloc(nW  * 2);
    __bf16* wvb = (__bf16*)alloc(nW  * 2);
    __bf16* owb = (__bf16*)alloc(nOW * 2);
    float*  qf  = (float*) alloc(nX  * 4);
    float*  kf  = (float*) alloc(nX  * 4);
    __bf16* vtb = (__bf16*)alloc(nX  * 2);   // v transposed per (b,h): [e, s]
    __bf16* sb  = (__bf16*)alloc(nX  * 2);   // final softmax, bf16
    __bf16* yb  = (__bf16*)alloc(nX  * 2);   // attention out in [B*S, H*D]

    // convert inputs to bf16
    cvt_f32_to_bf16<<<2048, 256, 0, stream>>>(x1, xb, nX / 4);
    cvt_f32_to_bf16<<<512,  256, 0, stream>>>(qw, wqb, nW / 4);
    cvt_f32_to_bf16<<<512,  256, 0, stream>>>(kw, wkb, nW / 4);
    cvt_f32_to_bf16<<<512,  256, 0, stream>>>(vw, wvb, nW / 4);
    cvt_f32_to_bf16<<<1024, 256, 0, stream>>>(ow, owb, nOW / 4);

    dim3 blk(256);
    dim3 gproj(Dd / 128, Sq / 128, Bsz * Hn);                 // (4,4,96)
    long sX = (long)Sq * Dd, sW = (long)Dd * Dd;

    // q = x @ Wq^T (f32) ; k = x @ Wk^T (f32) ; v = x @ Wv^T stored transposed (bf16)
    gemm_nt<0><<<gproj, blk, 0, stream>>>(xb, wqb, qf,  nullptr, Dd, Dd, Dd, Dd, sX, sW, sX, Hn);
    gemm_nt<0><<<gproj, blk, 0, stream>>>(xb, wkb, kf,  nullptr, Dd, Dd, Dd, Dd, sX, sW, sX, Hn);
    gemm_nt<1><<<gproj, blk, 0, stream>>>(xb, wvb, vtb, nullptr, Dd, Dd, Dd, Sq, sX, sW, sX, Hn);

    // softmax chain over S axis
    dim3 gsm(Dd / 256, Bsz * Hn);                              // (2,96)
    col_softmax<<<gsm, 256, 0, stream>>>(kf, nullptr, kf, 0);  // k_sm (in-place)
    col_softmax<<<gsm, 256, 0, stream>>>(qf, kf,      qf, 0);  // s = softmax(q) * k_sm
    col_softmax<<<gsm, 256, 0, stream>>>(qf, kf,      qf, 0);  // s = softmax(s) * k_sm
    col_softmax<<<gsm, 256, 0, stream>>>(qf, nullptr, sb, 1);  // s = softmax(s) -> bf16

    // attn = s @ v  (NT against vT), written straight into [b*S+s, h*D+e]
    gemm_nt<2><<<gproj, blk, 0, stream>>>(sb, vtb, yb, nullptr, Dd, Dd, Sq, HD,
                                          sX, (long)Dd * Sq, 0, 0);

    // out = y @ Wout^T + bias  (f32 output)
    dim3 gfin(HD / 128, (Bsz * Sq) / 128, 1);                  // (12,128,1)
    gemm_nt<3><<<gfin, blk, 0, stream>>>(yb, owb, (float*)d_out, bias, HD, HD, HD, HD,
                                         0, 0, 0, 0);
}
